// S4Backbone_25039659335760
// MI455X (gfx1250) — matrix operand order repaired and checked by reading, not compiled, not run on hardware
//
#include <hip/hip_runtime.h>
#include <hip/hip_bf16.h>

// ---------------- problem constants ----------------
#define N_LAYERS 4
#define Hdim 512
#define NSdim 32
#define Lseq 2048
#define Bbatch 8
#define Mrows (Bbatch * Lseq)      // 16384 token rows
#define TWO_H (2 * Hdim)           // 1024
#define LN_EPS 1e-5f

// ---------------- WMMA types ----------------
typedef __attribute__((ext_vector_type(16))) __bf16 v16bf;
typedef __attribute__((ext_vector_type(8)))  float  v8f;

union BF16Frag { unsigned short u[16]; v16bf v; };

__device__ __forceinline__ unsigned short f32_to_bf16(float f) {
  unsigned int x = __float_as_uint(f);
  unsigned int lsb = (x >> 16) & 1u;
  x += 0x7fffu + lsb;               // round-to-nearest-even
  return (unsigned short)(x >> 16);
}

__device__ __forceinline__ float gelu_exact(float x) {
  return 0.5f * x * (1.0f + erff(x * 0.70710678118654752f));
}

// -------- CDNA5 async global->LDS copy (ASYNCcnt-tracked DMA-style load) ----
// LDS[lds_off ..+15] = MEM[gsrc ..+15], per lane. Tracked by ASYNCcnt.
__device__ __forceinline__ void async_copy_b128(void* lds_dst, const void* gsrc) {
  unsigned lds_off = (unsigned)(size_t)lds_dst;   // low 32 bits of flat shared addr = LDS byte addr
  asm volatile("global_load_async_to_lds_b128 %0, %1, off"
               :: "v"(lds_off), "v"(gsrc)
               : "memory");
}
#define S_WAIT_ASYNCCNT(n) asm volatile("s_wait_asynccnt %0" :: "i"(n) : "memory")

// ---------------- simple utility kernels ----------------
__global__ __launch_bounds__(256) void copy_f32_kernel(const float* __restrict__ src,
                                                       float* __restrict__ dst, int n) {
  int i = blockIdx.x * 256 + threadIdx.x;
  if (i < n) dst[i] = src[i];
}

// Pre-swizzle W_out (all layers) into the exact v_wmma bf16 B-fragment layout.
// Linear output index o encodes:
//   [layer][nblk 0..7][half 0..1][ks 0..15][t 0..3][part 0..1][lane 0..31][jj 0..7]
// Fragment element v[j] (j = part*8+jj) of lane L is B[part*16 + (L>>4)*8 + jj][t*16 + (L&15)]
// of the 32x16 K-tile at k0 = ks*32, columns n = half*512 + nblk*64 + ...
__global__ __launch_bounds__(256) void swizzle_w_kernel(const float* __restrict__ W,
                                                        unsigned short* __restrict__ Wsw) {
  const int o     = blockIdx.x * 256 + threadIdx.x;   // 0 .. N_LAYERS*2^19-1
  const int jj    = o & 7;
  const int lane  = (o >> 3) & 31;
  const int part  = (o >> 8) & 1;
  const int t     = (o >> 9) & 3;
  const int ks    = (o >> 11) & 15;
  const int half  = (o >> 15) & 1;
  const int nblk  = (o >> 16) & 7;
  const int layer = o >> 19;
  const int k     = ks * 32 + part * 16 + (lane >> 4) * 8 + jj;
  const int ncol  = half * Hdim + nblk * 64 + t * 16 + (lane & 15);
  Wsw[o] = f32_to_bf16(W[((size_t)layer * Hdim + k) * TWO_H + ncol]);
}

// ---------------- S4D SSM recurrence + GELU ----------------
// One wave per (b,h); lane = SSM state n (NSdim == warpSize == 32).
// s[l] = w*s[l-1] + x[l],  y[l] = 2*Re(sum_n Ct_n * s_n[l]) + Dp*x[l]
// U(bf16) = gelu(y).
__global__ __launch_bounds__(256) void ssm_gelu_kernel(
    const float* __restrict__ X,        // (B,L,H)
    const float* __restrict__ log_dt,   // (H)   layer slice
    const float* __restrict__ A_real,   // (H,NS)
    const float* __restrict__ A_imag,   // (H,NS)
    const float* __restrict__ C_re,     // (H,NS)
    const float* __restrict__ C_im,     // (H,NS)
    const float* __restrict__ Dp,       // (H)
    unsigned short* __restrict__ Ubf)   // (B,L,H) bf16
{
  const int lane = threadIdx.x & 31;
  const int wid  = threadIdx.x >> 5;
  const int p    = blockIdx.x * 8 + wid;     // (b,h) pair, 0..B*H-1
  const int b    = p >> 9;                   // /H (H==512)
  const int h    = p & (Hdim - 1);
  const int n    = lane;

  // --- discretize: w = exp(dt*A), Ct = C * (w - 1)/A ---
  const float dt  = __expf(log_dt[h]);
  const float Are = -__expf(A_real[h * NSdim + n]);
  const float Aim = A_imag[h * NSdim + n];
  const float ar  = dt * Are, ai = dt * Aim;
  const float em  = __expf(ar);
  const float wr  = em * __cosf(ai);
  const float wi  = em * __sinf(ai);
  const float nr  = wr - 1.0f, ni = wi;               // w - 1
  const float den = Are * Are + Aim * Aim;
  const float qr  = (nr * Are + ni * Aim) / den;      // (w-1)/A
  const float qi  = (ni * Are - nr * Aim) / den;
  const float cr  = C_re[h * NSdim + n];
  const float ci  = C_im[h * NSdim + n];
  const float ctr = cr * qr - ci * qi;
  const float cti = cr * qi + ci * qr;
  const float dpv = Dp[h];

  float sr = 0.0f, si = 0.0f;
  const size_t base = (size_t)b * Lseq * Hdim + h;

  for (int l0 = 0; l0 < Lseq; l0 += 32) {
    // each lane fetches one timestep of this channel; broadcast inside chunk
    const float xmine = X[base + (size_t)(l0 + lane) * Hdim];
    float ysave = 0.0f;
    #pragma unroll 4
    for (int t = 0; t < 32; ++t) {
      const float xv = __shfl(xmine, t, 32);
      const float nsr = fmaf(wr, sr, fmaf(-wi, si, xv));
      const float nsi = fmaf(wr, si, wi * sr);
      sr = nsr; si = nsi;
      float contrib = fmaf(ctr, sr, -cti * si);
      // butterfly sum across 32 states
      #pragma unroll
      for (int off = 16; off >= 1; off >>= 1)
        contrib += __shfl_xor(contrib, off, 32);
      if (t == lane) ysave = contrib;
    }
    const float y = fmaf(dpv, xmine, 2.0f * ysave);
    Ubf[base + (size_t)(l0 + lane) * Hdim] = f32_to_bf16(gelu_exact(y));
  }
}

// ---------------- GEMM (bf16 WMMA) + bias + GLU + residual ----------------
// z = U @ W + b ; X += z[:, n] * sigmoid(z[:, n+H])
// Block tile: 128 rows x 64 column-pairs, 8 waves; wave w owns rows [16w,16w+16).
// A tile (128x32 bf16) is async-DMA'd into double-buffered LDS; B fragments are
// read directly from pre-swizzled global W (L2-resident: 192MB L2 >> 4MB W).
#define BLK_M 128
#define KSTEP 32
#define NKSTEPS (Hdim / KSTEP)     // 16

__global__ __launch_bounds__(256) void gemm_glu_res_kernel(
    const unsigned short* __restrict__ Ubf,   // (M, H) bf16
    const unsigned short* __restrict__ Wsw,   // swizzled (layer slice)
    const float* __restrict__ bias,           // (2H)
    float* __restrict__ X)                    // (M, H) += GLU
{
  __shared__ __align__(16) unsigned short As[2][BLK_M][KSTEP];   // 2 x 8 KB

  const int tid  = threadIdx.x;
  const int lane = tid & 31;
  const int wid  = tid >> 5;                 // 0..7
  const int m0   = blockIdx.x * BLK_M;
  const int nblk = blockIdx.y;               // 0..7
  const int n0   = nblk * 64;

  v8f acc1[4], acc2[4];
  #pragma unroll
  for (int t = 0; t < 4; ++t) { acc1[t] = 0; acc2[t] = 0; }

  // async-stage coordinates: 256 threads x 2 chunks cover 128x32 bf16 (8KB)
  const int sr_ = tid >> 2;                  // rows 0..63 (+64 for 2nd chunk)
  const int sc_ = (tid & 3) * 8;             // col 0,8,16,24

  // stage 0
  {
    const unsigned short* g = Ubf + (size_t)(m0 + sr_) * Hdim + sc_;
    async_copy_b128(&As[0][sr_][sc_],      g);
    async_copy_b128(&As[0][sr_ + 64][sc_], g + (size_t)64 * Hdim);
  }

  const unsigned short* wb = Wsw + (size_t)nblk * (2 * NKSTEPS * 4 * 512);

  for (int s = 0; s < NKSTEPS; ++s) {
    if (s + 1 < NKSTEPS) {
      const int k1 = (s + 1) * KSTEP;
      const unsigned short* g = Ubf + (size_t)(m0 + sr_) * Hdim + k1 + sc_;
      async_copy_b128(&As[(s + 1) & 1][sr_][sc_],      g);
      async_copy_b128(&As[(s + 1) & 1][sr_ + 64][sc_], g + (size_t)64 * Hdim);
      // prefetch next step's B fragments toward L2 as well
      __builtin_prefetch(wb + (((size_t)(s + 1)) * 4) * 512 + lane * 8, 0, 3);
      __builtin_prefetch(wb + (((size_t)NKSTEPS + s + 1) * 4) * 512 + lane * 8, 0, 3);
      S_WAIT_ASYNCCNT(2);     // current buffer's loads (issued earlier) complete
    } else {
      S_WAIT_ASYNCCNT(0);
    }
    __syncthreads();

    // A fragment from LDS: lane m<16 -> K{0..7,16..23}; lane m+16 -> K{8..15,24..31}
    BF16Frag af;
    {
      const unsigned short* ap = &As[s & 1][wid * 16 + (lane & 15)][0];
      const int kofs = (lane >> 4) * 8;
      *(uint4*)&af.u[0] = *(const uint4*)(ap + kofs);
      *(uint4*)&af.u[8] = *(const uint4*)(ap + 16 + kofs);
    }

    #pragma unroll
    for (int t = 0; t < 4; ++t) {
      // B fragments direct from swizzled global: 2 x b128 per fragment per lane
      const unsigned short* f0 = wb + (((size_t)0 * NKSTEPS + s) * 4 + t) * 512 + lane * 8;
      const unsigned short* f1 = wb + (((size_t)1 * NKSTEPS + s) * 4 + t) * 512 + lane * 8;
      BF16Frag b0, b1;
      *(uint4*)&b0.u[0] = *(const uint4*)(f0);
      *(uint4*)&b0.u[8] = *(const uint4*)(f0 + 256);
      *(uint4*)&b1.u[0] = *(const uint4*)(f1);
      *(uint4*)&b1.u[8] = *(const uint4*)(f1 + 256);

      acc1[t] = __builtin_amdgcn_wmma_f32_16x16x32_bf16(
          false, af.v, false, b0.v, (short)0, acc1[t], false, false);
      acc2[t] = __builtin_amdgcn_wmma_f32_16x16x32_bf16(
          false, af.v, false, b1.v, (short)0, acc2[t], false, false);
    }
    __syncthreads();
  }

  // epilogue: bias + GLU + residual. C/D layout: VGPR g -> row g + 8*(lane>>4), col = lane&15
  #pragma unroll
  for (int t = 0; t < 4; ++t) {
    const int col = n0 + t * 16 + (lane & 15);
    const float b1 = bias[col];
    const float b2 = bias[col + Hdim];
    #pragma unroll
    for (int g = 0; g < 8; ++g) {
      const int row = m0 + wid * 16 + g + 8 * (lane >> 4);
      const float z1 = acc1[t][g] + b1;
      const float z2 = acc2[t][g] + b2;
      const float sg = 1.0f / (1.0f + __expf(-z2));
      X[(size_t)row * Hdim + col] += z1 * sg;
    }
  }
}

// ---------------- final LayerNorm over H ----------------
__global__ __launch_bounds__(256) void layernorm_kernel(
    const float* __restrict__ X, const float* __restrict__ gvec,
    const float* __restrict__ bvec, float* __restrict__ out)
{
  const int row = blockIdx.x;
  const float* xr = X + (size_t)row * Hdim;
  float s = 0.0f, ss = 0.0f;
  for (int i = threadIdx.x; i < Hdim; i += 256) {
    const float v = xr[i];
    s += v; ss += v * v;
  }
  #pragma unroll
  for (int off = 16; off >= 1; off >>= 1) {
    s  += __shfl_xor(s,  off, 32);
    ss += __shfl_xor(ss, off, 32);
  }
  __shared__ float sh0[8], sh1[8];
  if ((threadIdx.x & 31) == 0) { sh0[threadIdx.x >> 5] = s; sh1[threadIdx.x >> 5] = ss; }
  __syncthreads();
  float ts = 0.0f, tss = 0.0f;
  #pragma unroll
  for (int w = 0; w < 8; ++w) { ts += sh0[w]; tss += sh1[w]; }
  const float mu  = ts * (1.0f / Hdim);
  const float var = tss * (1.0f / Hdim) - mu * mu;
  const float inv = rsqrtf(var + LN_EPS);
  for (int i = threadIdx.x; i < Hdim; i += 256)
    out[(size_t)row * Hdim + i] = (xr[i] - mu) * inv * gvec[i] + bvec[i];
}

// ---------------- host-side launch ----------------
extern "C" void kernel_launch(void* const* d_in, const int* in_sizes, int n_in,
                              void* d_out, int out_size, void* d_ws, size_t ws_size,
                              hipStream_t stream) {
  const float* x      = (const float*)d_in[0];
  const float* log_dt = (const float*)d_in[1];
  const float* A_real = (const float*)d_in[2];
  const float* A_imag = (const float*)d_in[3];
  const float* C_re   = (const float*)d_in[4];
  const float* C_im   = (const float*)d_in[5];
  const float* Dp     = (const float*)d_in[6];
  const float* W_out  = (const float*)d_in[7];
  const float* b_out  = (const float*)d_in[8];
  const float* ln_g   = (const float*)d_in[9];
  const float* ln_b   = (const float*)d_in[10];

  // workspace partition (bytes): X f32 | U bf16 | W bf16 (swizzled)
  const size_t XBYTES = (size_t)Mrows * Hdim * sizeof(float);          // 32 MB
  const size_t UBYTES = (size_t)Mrows * Hdim * sizeof(unsigned short); // 16 MB
  char* ws = (char*)d_ws;
  float*          Xbuf = (float*)ws;
  unsigned short* Ubf  = (unsigned short*)(ws + XBYTES);
  unsigned short* Wsw  = (unsigned short*)(ws + XBYTES + UBYTES);      // 4 MB

  const int nX = Mrows * Hdim;
  copy_f32_kernel<<<(nX + 255) / 256, 256, 0, stream>>>(x, Xbuf, nX);

  const int nW = N_LAYERS * Hdim * TWO_H;                              // 2^21
  swizzle_w_kernel<<<nW / 256, 256, 0, stream>>>(W_out, Wsw);

  const size_t W_LAYER = (size_t)Hdim * TWO_H;                         // swizzled elems/layer

  for (int i = 0; i < N_LAYERS; ++i) {
    ssm_gelu_kernel<<<(Bbatch * Hdim) / 8, 256, 0, stream>>>(
        Xbuf,
        log_dt + (size_t)i * Hdim,
        A_real + (size_t)i * Hdim * NSdim,
        A_imag + (size_t)i * Hdim * NSdim,
        C_re   + (size_t)i * Hdim * NSdim,
        C_im   + (size_t)i * Hdim * NSdim,
        Dp     + (size_t)i * Hdim,
        Ubf);

    dim3 grid(Mrows / BLK_M, 8);   // 128 x 8
    gemm_glu_res_kernel<<<grid, 256, 0, stream>>>(
        Ubf, Wsw + i * W_LAYER, b_out + (size_t)i * TWO_H, Xbuf);
  }

  layernorm_kernel<<<Mrows, 256, 0, stream>>>(Xbuf, ln_g, ln_b, (float*)d_out);
}
